// PalletizedLinear_81750407512100
// MI455X (gfx1250) — matrix operand order repaired and checked by reading, not compiled, not run on hardware
//
#include <hip/hip_runtime.h>

// PalletizedLinear for MI455X (gfx1250, wave32, WMMA).
// out[8,11008] = x[8,4096] @ dequant(W)^T + bias
// Roofline: index stream = 11008*4096*4 B = 180 MB read-once -> ~8 us at 23.3 TB/s.
// Everything else (lut 11.3 MB, x 128 KB, A-pack 128 KB) is L2-resident.
// Strategy: saturate HBM with NT b128 index loads; palette gather via LDS ds_load_u16;
// one v_wmma_f32_16x16x32_f16 per 16-col x 32-K tile; A fragments pre-packed to f16.

#define OUT_F 11008
#define IN_F  4096
#define GROUPSZ 128
#define NGROUPS_PER_ROW (IN_F / GROUPSZ)   // 32 palette groups per weight row

typedef __attribute__((ext_vector_type(16))) _Float16     v16h;
typedef __attribute__((ext_vector_type(8)))  float        v8f;
typedef __attribute__((ext_vector_type(8)))  unsigned int v8u;
typedef __attribute__((ext_vector_type(4)))  int          v4i;

#define NWAVES 8                    // waves per block, splitting K 8-way
#define KSEG   (IN_F / NWAVES)      // 512 K per wave
#define LUT_STRIDE 48               // staged LUT row stride (32B data + pad; 16B aligned, bank-spread)
#define LUT_AREA   (16 * LUT_STRIDE)        // 768 B per wave
#define RED_BASE   (NWAVES * LUT_AREA)      // 6144 B
#define LDS_BYTES  (RED_BASE + (NWAVES - 1) * 1024)

#define NCHUNKS (IN_F / 32)                 // 128 A fragments
#define APACK_BYTES ((size_t)NCHUNKS * 32 * 16 * sizeof(_Float16))   // 128 KB

// ---------------------------------------------------------------------------
// Pre-pass: convert x[8,4096] f32 -> f16 laid out in exact WMMA A-fragment
// order: apk[(chunk*32 + lane)*16 .. +15] = lane's 16 halves for that chunk.
// Rows 8..15 (zero padding of M) are stored as zeros.
// ---------------------------------------------------------------------------
__global__ __launch_bounds__(256)
void pack_x_a_fragments(const float* __restrict__ x, _Float16* __restrict__ apk)
{
    const int t    = blockIdx.x * 256 + threadIdx.x;   // 0 .. 4095
    const int lane = t & 31;
    const int kc   = (t >> 5) * 32;                    // chunk base K
    const int mrow = lane & 15;
    const int half = lane >> 4;

    float4 x0 = make_float4(0.f, 0.f, 0.f, 0.f);
    float4 x1 = x0, x2 = x0, x3 = x0;
    if (mrow < 8) {
        const float* xr = x + (size_t)mrow * IN_F + kc + half * 8;
        x0 = ((const float4*)xr)[0];
        x1 = ((const float4*)xr)[1];
        const float* xs = xr + 16;
        x2 = ((const float4*)xs)[0];
        x3 = ((const float4*)xs)[1];
    }
    v16h a;
    a[0]  = (_Float16)x0.x; a[1]  = (_Float16)x0.y;
    a[2]  = (_Float16)x0.z; a[3]  = (_Float16)x0.w;
    a[4]  = (_Float16)x1.x; a[5]  = (_Float16)x1.y;
    a[6]  = (_Float16)x1.z; a[7]  = (_Float16)x1.w;
    a[8]  = (_Float16)x2.x; a[9]  = (_Float16)x2.y;
    a[10] = (_Float16)x2.z; a[11] = (_Float16)x2.w;
    a[12] = (_Float16)x3.x; a[13] = (_Float16)x3.y;
    a[14] = (_Float16)x3.z; a[15] = (_Float16)x3.w;

    *(v8u*)(apk + (size_t)t * 16) = __builtin_bit_cast(v8u, a);
}

// ---------------------------------------------------------------------------
// Main kernel. PRE=true: A fragments come pre-packed (f16, 32 B/lane/chunk).
// PRE=false: fallback, convert x f32->f16 inline (if ws too small).
// ---------------------------------------------------------------------------
template<bool PRE>
__global__ __launch_bounds__(NWAVES * 32)
void pallet_linear_wmma_kernel(const float* __restrict__ x,
                               const _Float16* __restrict__ apk,
                               const int* __restrict__ widx,
                               const _Float16* __restrict__ lut,
                               const float* __restrict__ bias,
                               float* __restrict__ out)
{
    __shared__ __align__(16) unsigned char lds[LDS_BYTES];

    const int tid  = threadIdx.x;
    const int lane = tid & 31;
    const int wave = tid >> 5;
    const int half = lane >> 4;      // 0: lanes 0-15, 1: lanes 16-31
    const int mrow = lane & 15;
    const int n0   = blockIdx.x * 16;
    const int n    = n0 + mrow;      // this lane's fixed weight row / output column
    const int ks   = wave * KSEG;    // this wave's K segment start

    const _Float16* myRow = (const _Float16*)(&lds[wave * LUT_AREA + mrow * LUT_STRIDE]);

    v8f acc = {};   // 16x16 f32 C fragment (8 VGPRs)

    for (int kb0 = 0; kb0 < KSEG; kb0 += GROUPSZ) {
        const int kc0 = ks + kb0;

        // Stage the 16 palette rows for this 128-K block: lanes 0-15 pull
        // 32 B of fp16 LUT via two b128 loads -> LDS (48 B stride, bank-spread).
        if (lane < 16) {
            const size_t g = (size_t)(n0 + lane) * NGROUPS_PER_ROW + (size_t)(kc0 >> 7);
            const uint4* src = (const uint4*)(lut + g * 16);
            uint4 r0 = src[0];
            uint4 r1 = src[1];
            uint4* dst = (uint4*)(&lds[wave * LUT_AREA + lane * LUT_STRIDE]);
            dst[0] = r0;
            dst[1] = r1;
        }
        // Speculative prefetch of the next K block of this lane's index stream.
        __builtin_prefetch(widx + (size_t)n * IN_F + (kc0 + GROUPSZ) + half * 16, 0, 0);
        __syncthreads();

#pragma unroll
        for (int c = 0; c < GROUPSZ / 32; ++c) {
            const int kc = kc0 + c * 32;

            // ---- A fragment: X[0:16, kc:kc+32] as f16 (rows 8..15 zero pad).
            v16h a;
            if constexpr (PRE) {
                // Pre-packed in lane order: one 32 B load (2x global_load_b128), L2-hit.
                const v8u* ap = (const v8u*)(apk + ((size_t)(kc >> 5) * 32 + lane) * 16);
                a = __builtin_bit_cast(v16h, *ap);
            } else {
                float4 x0 = make_float4(0.f, 0.f, 0.f, 0.f);
                float4 x1 = x0, x2 = x0, x3 = x0;
                if (mrow < 8) {
                    const float* xr = x + (size_t)mrow * IN_F + kc + half * 8;
                    x0 = ((const float4*)xr)[0];
                    x1 = ((const float4*)xr)[1];
                    const float* xs = xr + 16;
                    x2 = ((const float4*)xs)[0];
                    x3 = ((const float4*)xs)[1];
                }
                a[0]  = (_Float16)x0.x; a[1]  = (_Float16)x0.y;
                a[2]  = (_Float16)x0.z; a[3]  = (_Float16)x0.w;
                a[4]  = (_Float16)x1.x; a[5]  = (_Float16)x1.y;
                a[6]  = (_Float16)x1.z; a[7]  = (_Float16)x1.w;
                a[8]  = (_Float16)x2.x; a[9]  = (_Float16)x2.y;
                a[10] = (_Float16)x2.z; a[11] = (_Float16)x2.w;
                a[12] = (_Float16)x3.x; a[13] = (_Float16)x3.y;
                a[14] = (_Float16)x3.z; a[15] = (_Float16)x3.w;
            }

            // ---- B fragment: dequantize W[n, kc+half*8 .. +7] and W[n, kc+16+half*8 .. +7].
            // 16 int32 indices (two runs of 8 consecutive, 16 apart) = 64 B: 4x NT global_load_b128
            // (read-once 180 MB stream; keep it out of L2's resident set),
            // then 16 ds_load_u16 palette gathers from the staged LDS row.
            const v4i* ip = (const v4i*)(widx + (size_t)n * IN_F + kc + half * 8);
            v4i i0 = __builtin_nontemporal_load(ip + 0);
            v4i i1 = __builtin_nontemporal_load(ip + 1);
            v4i i2 = __builtin_nontemporal_load(ip + 4);
            v4i i3 = __builtin_nontemporal_load(ip + 5);
            v16h b;
            b[0]  = myRow[i0.x]; b[1]  = myRow[i0.y];
            b[2]  = myRow[i0.z]; b[3]  = myRow[i0.w];
            b[4]  = myRow[i1.x]; b[5]  = myRow[i1.y];
            b[6]  = myRow[i1.z]; b[7]  = myRow[i1.w];
            b[8]  = myRow[i2.x]; b[9]  = myRow[i2.y];
            b[10] = myRow[i2.z]; b[11] = myRow[i2.w];
            b[12] = myRow[i3.x]; b[13] = myRow[i3.y];
            b[14] = myRow[i3.z]; b[15] = myRow[i3.w];

            // D = A x B + C  (16x16x32 f16 -> f32); uniform flow, EXEC all-1s.
            acc = __builtin_amdgcn_wmma_f32_16x16x32_f16(
                      /*neg_a=*/false, a, /*neg_b=*/false, b,
                      /*c_mod=*/(short)0, acc, /*reuse_a=*/false, /*reuse_b=*/false);
        }
    }

    // ---- Cross-wave K reduction through LDS, then bias + store by wave 0.
    float* red = (float*)(&lds[RED_BASE]);
    if (wave != 0) {
#pragma unroll
        for (int j = 0; j < 8; ++j)
            red[(wave - 1) * 256 + j * 32 + lane] = acc[j];   // [j][lane]: conflict-free
    }
    __syncthreads();
    if (wave == 0) {
#pragma unroll
        for (int j = 0; j < 8; ++j) {
            float s = acc[j];
#pragma unroll
            for (int w = 0; w < NWAVES - 1; ++w)
                s += red[w * 256 + j * 32 + lane];
            acc[j] = s;
        }
        // C layout: VGPR j, lanes 0-15 hold M=j (the 8 real rows), N = n0+lane.
        if (lane < 16) {
            const float bv = bias[n];
#pragma unroll
            for (int j = 0; j < 8; ++j)
                out[(size_t)j * OUT_F + n] = acc[j] + bv;
        }
    }
}

extern "C" void kernel_launch(void* const* d_in, const int* in_sizes, int n_in,
                              void* d_out, int out_size, void* d_ws, size_t ws_size,
                              hipStream_t stream) {
    const float*    x    = (const float*)d_in[0];
    const int*      widx = (const int*)d_in[1];
    const _Float16* lut  = (const _Float16*)d_in[2];
    const float*    bias = (const float*)d_in[3];
    float*          out  = (float*)d_out;
    _Float16*       apk  = (_Float16*)d_ws;

    dim3 grid(OUT_F / 16, 1, 1);     // 688 N-tiles
    dim3 block(NWAVES * 32, 1, 1);   // 8 waves splitting K

    if (ws_size >= APACK_BYTES) {
        pack_x_a_fragments<<<dim3(16), dim3(256), 0, stream>>>(x, apk);
        pallet_linear_wmma_kernel<true><<<grid, block, 0, stream>>>(x, apk, widx, lut, bias, out);
    } else {
        pallet_linear_wmma_kernel<false><<<grid, block, 0, stream>>>(x, apk, widx, lut, bias, out);
    }
}
